// MSABlock_5196910428487
// MI455X (gfx1250) — matrix operand order repaired and checked by reading, not compiled, not run on hardware
//
#include <hip/hip_runtime.h>
#include <math.h>

// ---------------------------------------------------------------------------
// MSA block (AF3-style) for gfx1250 / MI455X.
// All contractions -> v_wmma_f32_16x16x32_f16 (wave32, f32 accumulate).
// GEMM operands are pre-converted to f16 so global->LDS staging is pure
// 16-byte copies (async-to-LDS when available). LN/softmax/gating in f32.
// ---------------------------------------------------------------------------

typedef __attribute__((ext_vector_type(16))) _Float16 v16h;
typedef __attribute__((ext_vector_type(8)))  float    v8f;
typedef _Float16 f16;

#define TOK 256
#define MSA 128
#define CM  64
#define CZ  128

#define GBM 64
#define GBN 64
#define GBK 32
#define LDK 40        // padded halfword stride in LDS rows

#if defined(__HIP_DEVICE_COMPILE__) && __has_builtin(__builtin_amdgcn_global_load_async_to_lds_b128)
#define HAVE_ASYNC 1
#else
#define HAVE_ASYNC 0
#endif

__device__ __forceinline__ void copy16(const f16* g, f16* l) {
#if HAVE_ASYNC
  // clang prototype: void(v4i addrspace(1)*, v4i addrspace(3)*, imm i32, imm i32)
  typedef int v4i_t __attribute__((vector_size(16)));
  typedef __attribute__((address_space(1))) v4i_t gv4i_t;
  typedef __attribute__((address_space(3))) v4i_t lv4i_t;
  __builtin_amdgcn_global_load_async_to_lds_b128((gv4i_t*)g, (lv4i_t*)l, 0, 0);
#else
  *(float4*)l = *(const float4*)g;
#endif
}

__device__ __forceinline__ void async_fence() {
#if HAVE_ASYNC
#if __has_builtin(__builtin_amdgcn_s_wait_asynccnt)
  __builtin_amdgcn_s_wait_asynccnt(0);
#else
  asm volatile("s_wait_asynccnt 0x0" ::: "memory");
#endif
#endif
}

// ------------------------- generic batched WMMA GEMM ------------------------
// C = alpha * op(A) @ op(B) [+ bias] [+ C]
//   A,B are f16; C is f32 (cIsF16=0, beta allowed) or f16 (cIsF16=1).
//   ta: A stored [K,M] (lda = k-row stride), else [M,K]
//   tb: B stored [N,K] (ldb = n-row stride), else [K,N]
//   batch via blockIdx.z: off = (z / batchH) * s1 + (z % batchH) * s2
// Requirements (hold for every call site): M % 64 == 0, K % 32 == 0,
//   N % 8 == 0, lda/ldb % 8 == 0, 16B-aligned base pointers.
__global__ __launch_bounds__(128) void gemm_wmma_kernel(
    const f16* __restrict__ A, const f16* __restrict__ B,
    void* __restrict__ C, const float* __restrict__ bias,
    int M, int N, int K, int lda, int ldb, int ldc,
    int ta, int tb, float alpha, int beta, int cIsF16, int batchH,
    long long sA1, long long sA2, long long sB1, long long sB2,
    long long sC1, long long sC2)
{
  __shared__ f16 lA[GBM * LDK];
  __shared__ f16 lB[GBN * LDK];

  const int tid  = threadIdx.x;
  const int lane = tid & 31;
  const int wave = tid >> 5;
  const int m0   = blockIdx.x * GBM;
  const int n0   = blockIdx.y * GBN;
  const int bz   = blockIdx.z;
  const f16* Ab = A + (long long)(bz / batchH) * sA1 + (long long)(bz % batchH) * sA2;
  const f16* Bb = B + (long long)(bz / batchH) * sB1 + (long long)(bz % batchH) * sB2;
  const long long coff = (long long)(bz / batchH) * sC1 + (long long)(bz % batchH) * sC2;

  v8f acc[4];
  for (int t = 0; t < 4; ++t)
    for (int r = 0; r < 8; ++r) acc[t][r] = 0.0f;

  for (int k0 = 0; k0 < K; k0 += GBK) {
    __syncthreads();
    // ---- stage A tile: LDS layout lA[m][k] (64 x 32, padded) ----
    if (!ta) {
      // rows contiguous in k: 16B chunks, async-capable
      if (k0 + GBK < K)
        __builtin_prefetch(Ab + (long long)m0 * lda + (k0 + GBK), 0, 1);
      for (int t = 0; t < 2; ++t) {
        int task = tid + 128 * t;          // 256 tasks
        int r  = task >> 2;                // row 0..63
        int c8 = task & 3;                 // 8-half chunk of k
        copy16(Ab + (long long)(m0 + r) * lda + k0 + c8 * 8,
               lA + r * LDK + c8 * 8);
      }
    } else {
      // A[k][m]: vector global read, transposed scatter into LDS
      for (int t = 0; t < 2; ++t) {
        int task = tid + 128 * t;
        int kk = task >> 3;                // k row 0..31
        int c8 = task & 7;                 // 8-half chunk of m
        float4 v = *(const float4*)(Ab + (long long)(k0 + kk) * lda + m0 + c8 * 8);
        const f16* e = (const f16*)&v;
        for (int q = 0; q < 8; ++q) lA[(c8 * 8 + q) * LDK + kk] = e[q];
      }
    }
    // ---- stage B tile: LDS layout lB[n][k] (64 x 32, padded) ----
    if (tb) {
      // B[n][k]: rows contiguous in k, async-capable
      for (int t = 0; t < 2; ++t) {
        int task = tid + 128 * t;
        int r  = task >> 2;                // n row 0..63
        int c8 = task & 3;                 // k chunk
        if (n0 + r < N) {
          copy16(Bb + (long long)(n0 + r) * ldb + k0 + c8 * 8,
                 lB + r * LDK + c8 * 8);
        } else {
          float4 z; f16* e = (f16*)&z;
          for (int q = 0; q < 8; ++q) e[q] = (f16)0.0f;
          *(float4*)(lB + r * LDK + c8 * 8) = z;
        }
      }
    } else {
      // B[k][n]: vector global read, transposed scatter into LDS
      for (int t = 0; t < 2; ++t) {
        int task = tid + 128 * t;
        int kk = task >> 3;                // k row 0..31
        int c8 = task & 7;                 // 8-half chunk of n
        if (n0 + c8 * 8 < N) {
          float4 v = *(const float4*)(Bb + (long long)(k0 + kk) * ldb + n0 + c8 * 8);
          const f16* e = (const f16*)&v;
          for (int q = 0; q < 8; ++q) lB[(c8 * 8 + q) * LDK + kk] = e[q];
        } else {
          for (int q = 0; q < 8; ++q) lB[(c8 * 8 + q) * LDK + kk] = (f16)0.0f;
        }
      }
    }
    async_fence();
    __syncthreads();

    // A fragment: lane holds row (lane&15); K = 16*(i/8) + 8*(lane>>4) + (i%8)
    v16h af;
    {
      const f16* p = lA + (wave * 16 + (lane & 15)) * LDK + (lane >> 4) * 8;
      for (int i = 0; i < 8; ++i) { af[i] = p[i]; af[8 + i] = p[16 + i]; }
    }
    // B fragment: lane holds col (lane&15); K = 16*(i/8) + 8*(lane>>4) + (i%8)
    for (int t = 0; t < 4; ++t) {
      v16h bf;
      const f16* p = lB + (t * 16 + (lane & 15)) * LDK + (lane >> 4) * 8;
      for (int i = 0; i < 8; ++i) { bf[i] = p[i]; bf[8 + i] = p[16 + i]; }
      acc[t] = __builtin_amdgcn_wmma_f32_16x16x32_f16(
          false, af, false, bf, (short)0, acc[t], false, false);
    }
  }

  // C/D layout: VGPR r -> row r + 8*(lane>>4), col = lane&15
  for (int t = 0; t < 4; ++t) {
    int gc = n0 + t * 16 + (lane & 15);
    if (gc >= N) continue;
    for (int r = 0; r < 8; ++r) {
      int gr = m0 + wave * 16 + (lane >> 4) * 8 + r;
      if (gr >= M) continue;
      long long ci = (long long)gr * ldc + gc;
      float v = alpha * acc[t][r];
      if (bias) v += bias[gc];
      if (cIsF16) {
        ((f16*)C + coff)[ci] = (f16)v;
      } else {
        float* Cp = (float*)C + coff;
        if (beta) v += Cp[ci];
        Cp[ci] = v;
      }
    }
  }
}

// --------------------------- naive GEMM for N < 16 --------------------------
__global__ void gemm_small_kernel(const f16* __restrict__ A,
                                  const float* __restrict__ B,
                                  float* __restrict__ C,
                                  long long M, int N, int K)
{
  long long idx = (long long)blockIdx.x * blockDim.x + threadIdx.x;
  long long stride = (long long)gridDim.x * blockDim.x;
  long long total = M * N;
  for (; idx < total; idx += stride) {
    int col = (int)(idx % N);
    long long row = idx / N;
    float s = 0.0f;
    const f16* a = A + row * K;
    for (int k = 0; k < K; ++k) s += (float)a[k] * B[(long long)k * N + col];
    C[idx] = s;
  }
}

// --------------------------------- layernorm --------------------------------
// f32 in, f16 out (consumed only as GEMM operands)
__global__ __launch_bounds__(128) void layernorm_kernel(
    const float* __restrict__ x, const float* __restrict__ g,
    const float* __restrict__ b, f16* __restrict__ y,
    long long rows, int C)
{
  long long row = (long long)blockIdx.x * 4 + (threadIdx.x >> 5);
  if (row >= rows) return;
  int lane = threadIdx.x & 31;
  const float* xr = x + row * C;
  float s = 0.0f, s2 = 0.0f;
  for (int c = lane; c < C; c += 32) { float v = xr[c]; s += v; s2 += v * v; }
  for (int o = 16; o > 0; o >>= 1) {
    s  += __shfl_xor(s,  o, 32);
    s2 += __shfl_xor(s2, o, 32);
  }
  float mu  = s / C;
  float var = s2 / C - mu * mu;
  float inv = rsqrtf(var + 1e-5f);
  f16* yr = y + row * C;
  for (int c = lane; c < C; c += 32)
    yr[c] = (f16)((xr[c] - mu) * inv * g[c] + b[c]);
}

// ------------------------------- elementwise --------------------------------
__device__ __forceinline__ float sigf(float x) { return 1.0f / (1.0f + expf(-x)); }

__global__ void cvt_f32_f16_kernel(const float* __restrict__ s, f16* __restrict__ d, long long n)
{
  long long i = (long long)blockIdx.x * blockDim.x + threadIdx.x;
  long long st = (long long)gridDim.x * blockDim.x;
  for (; i < n; i += st) d[i] = (f16)s[i];
}

// a = mask[idx/C] * sigmoid(a) * bv   (f32 in place)
__global__ void ew_sig_mul_mask_kernel(float* a, const float* bv,
                                       const float* mask, long long n, int C)
{
  long long i = (long long)blockIdx.x * blockDim.x + threadIdx.x;
  long long st = (long long)gridDim.x * blockDim.x;
  for (; i < n; i += st) a[i] = mask[i / C] * sigf(a[i]) * bv[i];
}

// o16 = silu(a) * b
__global__ void ew_silu_mul_f16_kernel(const float* a, const float* b,
                                       f16* o, long long n)
{
  long long i = (long long)blockIdx.x * blockDim.x + threadIdx.x;
  long long st = (long long)gridDim.x * blockDim.x;
  for (; i < n; i += st) { float x = a[i]; o[i] = (f16)(x * sigf(x) * b[i]); }
}

// o16 = sigmoid(g) * x
__global__ void ew_sig_mul_f16_kernel(const float* g, const float* x,
                                      f16* o, long long n)
{
  long long i = (long long)blockIdx.x * blockDim.x + threadIdx.x;
  long long st = (long long)gridDim.x * blockDim.x;
  for (; i < n; i += st) o[i] = (f16)(sigf(g[i]) * x[i]);
}

// z += sigmoid(g) * t
__global__ void ew_gated_add_kernel(float* z, const float* g, const float* t, long long n)
{
  long long i = (long long)blockIdx.x * blockDim.x + threadIdx.x;
  long long st = (long long)gridDim.x * blockDim.x;
  for (; i < n; i += st) z[i] += sigf(g[i]) * t[i];
}

// ------------------------------ layout packers ------------------------------
// dst16[c*65536 + k*256 + i] = src[(k*256+i)*128 + c]   (trans=0, incoming)
//                            = src[(i*256+k)*128 + c]   (trans=1, outgoing)
__global__ void pack_cm_kernel(const float* __restrict__ src, f16* __restrict__ dst, int trans)
{
  long long idx = (long long)blockIdx.x * blockDim.x + threadIdx.x;
  long long st  = (long long)gridDim.x * blockDim.x;
  for (; idx < (long long)CZ * TOK * TOK; idx += st) {
    int c = (int)(idx >> 16);
    int rem = (int)(idx & 65535);
    int k = rem >> 8, i = rem & 255;
    long long s = trans ? ((long long)(i * TOK + k) * CZ + c)
                        : ((long long)(k * TOK + i) * CZ + c);
    dst[idx] = (f16)src[s];
  }
}

// dst[(i*256+j)*128 + c] = src[c*65536 + i*256 + j]  (f32 -> f32, feeds LN)
__global__ void unpack_cm_kernel(const float* __restrict__ src, float* __restrict__ dst)
{
  long long idx = (long long)blockIdx.x * blockDim.x + threadIdx.x;
  long long st  = (long long)gridDim.x * blockDim.x;
  for (; idx < (long long)CZ * TOK * TOK; idx += st) {
    int c = (int)(idx & 127);
    long long t = idx >> 7;
    int j = (int)(t & 255), i = (int)(t >> 8);
    dst[idx] = src[(long long)c * 65536 + i * TOK + j];
  }
}

// P16[(il*256+j)*1024 + c*32+d] = O[(il*32+c)*8192 + j*32+d]
__global__ void opm_pack_kernel(const float* __restrict__ O, f16* __restrict__ P)
{
  long long idx = (long long)blockIdx.x * blockDim.x + threadIdx.x;
  long long st  = (long long)gridDim.x * blockDim.x;
  for (; idx < (long long)8192 * 1024; idx += st) {
    int kk = (int)(idx & 1023);
    long long r = idx >> 10;
    int j = (int)(r & 255), il = (int)(r >> 8);
    int c = kk >> 5, d = kk & 31;
    P[idx] = (f16)O[(long long)(il * 32 + c) * 8192 + j * 32 + d];
  }
}

// wp16[h*65536 + i*256 + j] = w[(i*256+j)*8 + h]
__global__ void pack_w_kernel(const float* __restrict__ w, f16* __restrict__ wp)
{
  long long idx = (long long)blockIdx.x * blockDim.x + threadIdx.x;
  long long st  = (long long)gridDim.x * blockDim.x;
  for (; idx < (long long)8 * 65536; idx += st) {
    int h = (int)(idx >> 16);
    int rem = (int)(idx & 65535);
    wp[idx] = (f16)w[(long long)rem * 8 + h];
  }
}

// in-place transpose of z over (i,j)
__global__ void transpose_inplace_kernel(float* z)
{
  long long idx = (long long)blockIdx.x * blockDim.x + threadIdx.x;
  long long st  = (long long)gridDim.x * blockDim.x;
  for (; idx < (long long)TOK * TOK * CZ; idx += st) {
    int c = (int)(idx & 127);
    long long t = idx >> 7;
    int j = (int)(t & 255), i = (int)(t >> 8);
    if (i < j) {
      long long b = ((long long)(j * TOK + i) * CZ) + c;
      float va = z[idx], vb = z[b];
      z[idx] = vb; z[b] = va;
    }
  }
}

// -------------------------------- softmaxes ---------------------------------
// mpwa: softmax over j of b[(i*256+j)*8 + h]; grid (i=256, h=8)
__global__ __launch_bounds__(256) void softmax_j_kernel(float* b)
{
  __shared__ float red[256];
  int i = blockIdx.x, h = blockIdx.y, j = threadIdx.x;
  long long idx = ((long long)(i * TOK + j)) * 8 + h;
  float v = b[idx];
  red[j] = v; __syncthreads();
  for (int s = 128; s > 0; s >>= 1) { if (j < s) red[j] = fmaxf(red[j], red[j + s]); __syncthreads(); }
  float mx = red[0]; __syncthreads();
  float e = expf(v - mx);
  red[j] = e; __syncthreads();
  for (int s = 128; s > 0; s >>= 1) { if (j < s) red[j] += red[j + s]; __syncthreads(); }
  b[idx] = e / red[0];
}

// attention: read f32 scores, apply scale+bias+mask, softmax over kk, write f16
__global__ __launch_bounds__(256) void softmax_att_kernel(
    const float* __restrict__ S, const float* __restrict__ bias,
    const float* __restrict__ mask, int i0, int maskT, f16* __restrict__ S16)
{
  __shared__ float red[256];
  int jj = blockIdx.x, h = blockIdx.y, il = blockIdx.z;
  int kk = threadIdx.x;
  int i = i0 + il;
  long long base = (((long long)il * 4 + h) * TOK + jj) * TOK;
  float mval = maskT ? mask[(long long)kk * TOK + i] : mask[(long long)i * TOK + kk];
  float v = S[base + kk] * 0.17677669529663687f
          + bias[(long long)(jj * TOK + kk) * 4 + h]
          + (mval - 1.0f) * 1e9f;
  red[kk] = v; __syncthreads();
  for (int s = 128; s > 0; s >>= 1) { if (kk < s) red[kk] = fmaxf(red[kk], red[kk + s]); __syncthreads(); }
  float mx = red[0]; __syncthreads();
  float e = expf(v - mx);
  red[kk] = e; __syncthreads();
  for (int s = 128; s > 0; s >>= 1) { if (kk < s) red[kk] += red[kk + s]; __syncthreads(); }
  S16[base + kk] = (f16)(e / red[0]);
}

// ------------------------------ host-side glue ------------------------------
static void gemm(hipStream_t s, const f16* A, const f16* B, void* C,
                 const float* bias, int M, int N, int K,
                 int lda, int ldb, int ldc, int ta, int tb,
                 float alpha, int beta, int cIsF16,
                 int batch = 1, int batchH = 1,
                 long long sA1 = 0, long long sA2 = 0,
                 long long sB1 = 0, long long sB2 = 0,
                 long long sC1 = 0, long long sC2 = 0)
{
  dim3 g((M + GBM - 1) / GBM, (N + GBN - 1) / GBN, batch);
  gemm_wmma_kernel<<<g, 128, 0, s>>>(A, B, C, bias, M, N, K, lda, ldb, ldc,
                                     ta, tb, alpha, beta, cIsF16, batchH,
                                     sA1, sA2, sB1, sB2, sC1, sC2);
}

static void layernorm(hipStream_t s, const float* x, const float* g,
                      const float* b, f16* y, long long rows, int C)
{
  layernorm_kernel<<<dim3((unsigned)((rows + 3) / 4)), 128, 0, s>>>(x, g, b, y, rows, C);
}

#define EWGRID dim3(2048), dim3(256)
#define F16P(p) ((f16*)(p))

extern "C" void kernel_launch(void* const* d_in, const int* in_sizes, int n_in,
                              void* d_out, int out_size, void* d_ws, size_t ws_size,
                              hipStream_t stream)
{
  (void)in_sizes; (void)n_in; (void)out_size; (void)ws_size;
#define IN(i) ((const float*)d_in[i])

  const long long MSZ = (long long)MSA * TOK * CM;   // 2,097,152
  const long long ZSZ = (long long)TOK * TOK * CZ;   // 8,388,608

  float* mW = (float*)d_out;          // working m  [128,256,64]
  float* zW = mW + MSZ;               // working z  [256,256,128]
  (void)hipMemcpyAsync(mW, d_in[0], MSZ * sizeof(float), hipMemcpyDeviceToDevice, stream);
  (void)hipMemcpyAsync(zW, d_in[1], ZSZ * sizeof(float), hipMemcpyDeviceToDevice, stream);
  const float* mask = IN(2);

  // workspace: six 8M-f32 slabs (raw storage, f32 or f16 views), small + weights
  float* ws = (float*)d_ws;
  const long long M1 = 1ll << 20;
  float* S1 = ws;
  float* S2 = ws +  8 * M1;
  float* S3 = ws + 16 * M1;
  float* S4 = ws + 24 * M1;
  float* S5 = ws + 32 * M1;
  float* S6 = ws + 40 * M1;
  float* SM = ws + 48 * M1;           // 2M f32 small region
  f16*   WB = (f16*)(ws + 50 * M1);   // f16 weight copies (<= 2M halves)

  long long woff = 0;
  auto cvtW = [&](const float* src, long long n) -> f16* {
    f16* dst = WB + woff;
    cvt_f32_f16_kernel<<<dim3(256), dim3(256), 0, stream>>>(src, dst, n);
    woff += n;
    return dst;
  };
  f16* Wopm_a = cvtW(IN(5), 64 * 32);
  f16* Wopm_b = cvtW(IN(6), 64 * 32);
  f16* Wopm_o = cvtW(IN(7), 1024 * 128);
  f16* Wmp_v  = cvtW(IN(13), 64 * 256);
  f16* Wmp_g  = cvtW(IN(15), 64 * 256);
  f16* Wmp_o  = cvtW(IN(16), 256 * 64);
  f16* Wtm_a  = cvtW(IN(19), 64 * 256);
  f16* Wtm_b  = cvtW(IN(20), 64 * 256);
  f16* Wtm_o  = cvtW(IN(21), 256 * 64);
  f16* Wtri[2][6]; // Wa, Wag, Wb, Wbg, Wg, Wo
  for (int w = 0; w < 2; ++w) {
    int pb = (w == 0) ? 22 : 32;
    Wtri[w][0] = cvtW(IN(pb + 2), 128 * 128);
    Wtri[w][1] = cvtW(IN(pb + 3), 128 * 128);
    Wtri[w][2] = cvtW(IN(pb + 4), 128 * 128);
    Wtri[w][3] = cvtW(IN(pb + 5), 128 * 128);
    Wtri[w][4] = cvtW(IN(pb + 6), 128 * 128);
    Wtri[w][5] = cvtW(IN(pb + 9), 128 * 128);
  }
  f16* Watt[2][5]; // Wq, Wk, Wv, Wg, Wo
  for (int w = 0; w < 2; ++w) {
    int pb = (w == 0) ? 42 : 50;
    Watt[w][0] = cvtW(IN(pb + 2), 128 * 128);
    Watt[w][1] = cvtW(IN(pb + 3), 128 * 128);
    Watt[w][2] = cvtW(IN(pb + 4), 128 * 128);
    Watt[w][3] = cvtW(IN(pb + 6), 128 * 128);
    Watt[w][4] = cvtW(IN(pb + 7), 128 * 128);
  }
  f16* Wtz_a = cvtW(IN(60), 128 * 512);
  f16* Wtz_b = cvtW(IN(61), 128 * 512);
  f16* Wtz_o = cvtW(IN(62), 512 * 128);

  // ======================= 1) z += OuterProductMean(m) ======================
  layernorm(stream, mW, IN(3), IN(4), F16P(SM), 32768, CM);
  gemm(stream, F16P(SM), Wopm_a, S1, nullptr, 32768, 32, 64, 64, 32, 32, 0, 0, 1.0f, 0, 1);
  gemm(stream, F16P(SM), Wopm_b, S2, nullptr, 32768, 32, 64, 64, 32, 32, 0, 0, 1.0f, 0, 1);
  for (int ic = 0; ic < 8; ++ic) {
    int ib = ic * 32;
    // O[(i_loc,c),(j,d)] = sum_s a16[s,(ib+i_loc),c] * b16[s,j,d]  (TN, K=128)
    gemm(stream, F16P(S1) + ib * 32, F16P(S2), S3, nullptr,
         1024, 8192, 128, 8192, 8192, 8192, 1, 0, 1.0f, 0, 0);
    opm_pack_kernel<<<EWGRID, 0, stream>>>(S3, F16P(S4));
    gemm(stream, F16P(S4), Wopm_o, zW + (long long)ib * TOK * CZ, IN(8),
         8192, 128, 1024, 1024, 128, 128, 0, 0, 1.0f / 128.0f, 1, 0);
  }

  // ==================== 2) m += MSAPairWeightedAveraging ====================
  layernorm(stream, mW, IN(9),  IN(10), F16P(SM), 32768, CM);       // mn16
  layernorm(stream, zW, IN(11), IN(12), F16P(S1), 65536, CZ);       // zn16
  gemm(stream, F16P(SM), Wmp_v, S2, nullptr, 32768, 256, 64, 64, 256, 256, 0, 0, 1.0f, 0, 1); // v16
  gemm_small_kernel<<<EWGRID, 0, stream>>>(F16P(S1), IN(14), SM + M1, 65536, 8, CZ);          // b[i,j,h]
  gemm(stream, F16P(SM), Wmp_g, S3, nullptr, 32768, 256, 64, 64, 256, 256, 0, 0, 1.0f, 0, 0); // g f32
  softmax_j_kernel<<<dim3(256, 8), 256, 0, stream>>>(SM + M1);
  f16* wp16 = (f16*)(SM + M1 + M1 / 2);
  pack_w_kernel<<<EWGRID, 0, stream>>>(SM + M1, wp16);
  // wv[s,i,h,c] = sum_j wp16[h][i][j] * v16[s,j,h,c]   batch=(s,h)=1024
  gemm(stream, wp16, F16P(S2), S4, nullptr, 256, 32, 256, 256, 256, 256, 0, 0, 1.0f, 0, 0,
       1024, 8, 0, 65536, 65536, 32, 65536, 32);
  ew_sig_mul_f16_kernel<<<EWGRID, 0, stream>>>(S3, S4, F16P(S5), (long long)32768 * 256);
  gemm(stream, F16P(S5), Wmp_o, mW, nullptr, 32768, 64, 256, 256, 64, 64, 0, 0, 1.0f, 1, 0);

  // ========================= 3) m += Transition(m) ==========================
  layernorm(stream, mW, IN(17), IN(18), F16P(S1), 32768, CM);
  gemm(stream, F16P(S1), Wtm_a, S2, nullptr, 32768, 256, 64, 64, 256, 256, 0, 0, 1.0f, 0, 0);
  gemm(stream, F16P(S1), Wtm_b, S3, nullptr, 32768, 256, 64, 64, 256, 256, 0, 0, 1.0f, 0, 0);
  ew_silu_mul_f16_kernel<<<EWGRID, 0, stream>>>(S2, S3, F16P(S4), (long long)32768 * 256);
  gemm(stream, F16P(S4), Wtm_o, mW, nullptr, 32768, 64, 256, 256, 64, 64, 0, 0, 1.0f, 1, 0);

  // ==================== 4,5) z += TriangleMultiplication ====================
  for (int w = 0; w < 2; ++w) {
    int pb = (w == 0) ? 22 : 32;            // tmo (outgoing) then tmi (incoming)
    int outgoing = (w == 0) ? 1 : 0;
    layernorm(stream, zW, IN(pb), IN(pb + 1), F16P(S1), 65536, CZ);
    gemm(stream, F16P(S1), Wtri[w][1], S2, nullptr, 65536, CZ, CZ, CZ, CZ, CZ, 0, 0, 1.0f, 0, 0); // ag
    gemm(stream, F16P(S1), Wtri[w][0], S3, nullptr, 65536, CZ, CZ, CZ, CZ, CZ, 0, 0, 1.0f, 0, 0); // av
    ew_sig_mul_mask_kernel<<<EWGRID, 0, stream>>>(S2, S3, mask, ZSZ, CZ);                         // a in S2
    gemm(stream, F16P(S1), Wtri[w][3], S3, nullptr, 65536, CZ, CZ, CZ, CZ, CZ, 0, 0, 1.0f, 0, 0); // bg
    gemm(stream, F16P(S1), Wtri[w][2], S4, nullptr, 65536, CZ, CZ, CZ, CZ, CZ, 0, 0, 1.0f, 0, 0); // bv
    ew_sig_mul_mask_kernel<<<EWGRID, 0, stream>>>(S3, S4, mask, ZSZ, CZ);                         // b in S3
    pack_cm_kernel<<<EWGRID, 0, stream>>>(S2, F16P(S4), outgoing);   // a_cm16
    pack_cm_kernel<<<EWGRID, 0, stream>>>(S3, F16P(S5), outgoing);   // b_cm16
    // X_c[i][j] = sum_k a_cm[c][k][i] * b_cm[c][k][j]  (128-way batched TN)
    gemm(stream, F16P(S4), F16P(S5), S6, nullptr, 256, 256, 256, 256, 256, 256, 1, 0, 1.0f, 0, 0,
         128, 1, 65536, 0, 65536, 0, 65536, 0);
    unpack_cm_kernel<<<EWGRID, 0, stream>>>(S6, S2);                 // x f32
    layernorm(stream, S2, IN(pb + 7), IN(pb + 8), F16P(S4), 65536, CZ); // xn16
    gemm(stream, F16P(S1), Wtri[w][4], S5, nullptr, 65536, CZ, CZ, CZ, CZ, CZ, 0, 0, 1.0f, 0, 0); // g
    gemm(stream, F16P(S4), Wtri[w][5], S3, nullptr, 65536, CZ, CZ, CZ, CZ, CZ, 0, 0, 1.0f, 0, 0); // t
    ew_gated_add_kernel<<<EWGRID, 0, stream>>>(zW, S5, S3, ZSZ);
  }

  // ====================== 6,7) z += TriangleAttention =======================
  for (int w = 0; w < 2; ++w) {
    int pb = (w == 0) ? 42 : 50;            // tas then tae
    int maskT = w;
    if (w == 1) transpose_inplace_kernel<<<EWGRID, 0, stream>>>(zW);   // z -> zt
    layernorm(stream, zW, IN(pb), IN(pb + 1), F16P(S1), 65536, CZ);
    gemm(stream, F16P(S1), Watt[w][0], S2, nullptr, 65536, CZ, CZ, CZ, CZ, CZ, 0, 0, 1.0f, 0, 1); // q16
    gemm(stream, F16P(S1), Watt[w][1], S3, nullptr, 65536, CZ, CZ, CZ, CZ, CZ, 0, 0, 1.0f, 0, 1); // k16
    gemm(stream, F16P(S1), Watt[w][2], S4, nullptr, 65536, CZ, CZ, CZ, CZ, CZ, 0, 0, 1.0f, 0, 1); // v16
    gemm_small_kernel<<<EWGRID, 0, stream>>>(F16P(S1), IN(pb + 5), SM, 65536, 4, CZ);             // bias
    gemm(stream, F16P(S1), Watt[w][3], S5, nullptr, 65536, CZ, CZ, CZ, CZ, CZ, 0, 0, 1.0f, 0, 0); // g f32
    f16* S16 = F16P(S2) + 8 * M1;      // upper half of S2 (q16 uses lower 16MB)
    for (int ic = 0; ic < 8; ++ic) {
      int i0 = ic * 32;
      // S[j][kk] = sum_c q16[i,j,h,c] * k16[i,kk,h,c]   batch=(i_loc,h)=128
      gemm(stream, F16P(S2) + (long long)i0 * 32768, F16P(S3) + (long long)i0 * 32768, S6, nullptr,
           256, 256, 32, 128, 128, 256, 0, 1, 1.0f, 0, 0,
           128, 4, 32768, 32, 32768, 32, 262144, 65536);
      softmax_att_kernel<<<dim3(256, 4, 32), 256, 0, stream>>>(S6, SM, mask, i0, maskT, S16);
      // o[j][c] = sum_kk S16[j,kk] * v16[i,kk,h,c]  -> o f32 in S1
      gemm(stream, S16, F16P(S4) + (long long)i0 * 32768, S1 + (long long)i0 * 32768, nullptr,
           256, 32, 256, 256, 128, 128, 0, 0, 1.0f, 0, 0,
           128, 4, 262144, 65536, 32768, 32, 32768, 32);
    }
    ew_sig_mul_f16_kernel<<<EWGRID, 0, stream>>>(S5, S1, F16P(S6), ZSZ);   // go16
    gemm(stream, F16P(S6), Watt[w][4], zW, nullptr, 65536, CZ, CZ, CZ, CZ, CZ, 0, 0, 1.0f, 1, 0);
    if (w == 1) transpose_inplace_kernel<<<EWGRID, 0, stream>>>(zW);       // zt -> z
  }

  // ========================= 8) z += Transition(z) ==========================
  layernorm(stream, zW, IN(58), IN(59), F16P(S1), 65536, CZ);
  for (int rc = 0; rc < 4; ++rc) {          // 16384-row chunks (hidden=512)
    long long r0 = (long long)rc * 16384;
    const f16* xn = F16P(S1) + r0 * CZ;
    gemm(stream, xn, Wtz_a, S2, nullptr, 16384, 512, CZ, CZ, 512, 512, 0, 0, 1.0f, 0, 0);
    gemm(stream, xn, Wtz_b, S3, nullptr, 16384, 512, CZ, CZ, 512, 512, 0, 0, 1.0f, 0, 0);
    ew_silu_mul_f16_kernel<<<EWGRID, 0, stream>>>(S2, S3, F16P(S4), (long long)16384 * 512);
    gemm(stream, F16P(S4), Wtz_o, zW + r0 * CZ, nullptr, 16384, CZ, 512, 512, CZ, CZ, 0, 0, 1.0f, 1, 0);
  }
#undef IN
}
